// ArcFaceLoss_76381698392241
// MI455X (gfx1250) — matrix operand order repaired and checked by reading, not compiled, not run on hardware
//
#include <hip/hip_runtime.h>
#include <hip/hip_bf16.h>
#include <math.h>

// ---------------------------------------------------------------------------
// ArcFace loss for MI455X (gfx1250, wave32, WMMA).
//
//   e[512,512] f32, w[100000,512] f32, labels[512] i32 -> scalar mean NLL
//
// Pipeline:
//   K0: L2-normalize e -> e16 (f16)                       [tiny]
//   K1: fused GEMM: cos = (e16 . f16(w)) * rsqrt(ssq_w),  v_wmma_f32_16x16x32_f16
//       + per-(row, 16-class chunk) partial sum of exp(64*cos - M0)
//   K2: per-row: reduce 6250 partials, label-column margin correction, nll
//   K3: mean over 512 rows
// ---------------------------------------------------------------------------

typedef _Float16 v16h __attribute__((ext_vector_type(16)));
typedef _Float16 v8h  __attribute__((ext_vector_type(8)));
typedef float    v8f  __attribute__((ext_vector_type(8)));

#define EMB_DIM      512
#define BATCH        512
#define NUM_CLASSES  100000
#define NCHUNKS      6250          // 100000 / 16
#define GEMM_BLOCKS  3125          // 100000 / 32
#define SSCALE       64.0f
#define M0_SHIFT     20.0f         // fixed logsumexp shift (logits in [-64,64], typ. |.|<16)
#define CLIPV        (1.0f - 1e-7f)

#define ARC_COS_M  0.87758256189037271612f   // cos(0.5)
#define ARC_SIN_M  0.47942553860420300027f   // sin(0.5)
#define ARC_TH    (-0.87758256189037271612f) // cos(pi - 0.5)
#define ARC_MM     0.23971276930210150013f   // sin(pi - 0.5) * 0.5

// ---------------------------------------------------------------------- K0 --
__global__ __launch_bounds__(256)
void norm_e_kernel(const float* __restrict__ e, _Float16* __restrict__ e16) {
  const int lane = threadIdx.x & 31;
  const int row  = blockIdx.x * 8 + (threadIdx.x >> 5);
  const float* er = e + (size_t)row * EMB_DIM;
  float vals[16];
  float ssq = 0.0f;
#pragma unroll
  for (int j = 0; j < 16; ++j) {
    float x = er[lane + j * 32];
    vals[j] = x;
    ssq = fmaf(x, x, ssq);
  }
#pragma unroll
  for (int m = 1; m < 32; m <<= 1) ssq += __shfl_xor(ssq, m, 32);
  const float sc = 1.0f / fmaxf(sqrtf(ssq), 1e-12f);
  _Float16* orow = e16 + (size_t)row * EMB_DIM;
#pragma unroll
  for (int j = 0; j < 16; ++j) orow[lane + j * 32] = (_Float16)(vals[j] * sc);
}

// ---------------------------------------------------------------------- K1 --
// Block: 256 threads = 8 waves. Covers N=32 classes x M=512 rows.
//   wave w: nw = w>>2 (class half), mw = w&3 (128-row strip, 8 M-tiles).
// Per k-step: 1 B-fragment (fp32 w -> f16, + sumsq), 8 A-fragments, 8 WMMAs.
__global__ __launch_bounds__(256)
void arcface_gemm_kernel(const float* __restrict__ w,
                         const _Float16* __restrict__ e16,
                         float* __restrict__ part) {
  const int tid  = threadIdx.x;
  const int lane = tid & 31;
  const int wv   = tid >> 5;
  const int nw   = wv >> 2;          // 0..1
  const int mw   = wv & 3;           // 0..3
  const int l15  = lane & 15;
  const int hi   = lane >> 4;        // lane group (0: lanes 0-15, 1: 16-31)

  const int cls     = blockIdx.x * 32 + nw * 16 + l15;  // this lane's B column
  const int rowbase = mw * 128;

  const float* wrow = w + (size_t)cls * EMB_DIM;

  v8f c[8];
#pragma unroll
  for (int i = 0; i < 8; ++i)
    c[i] = (v8f){0.f, 0.f, 0.f, 0.f, 0.f, 0.f, 0.f, 0.f};

  float ssq = 0.0f;   // sumsq of this lane's f16(w) values (half of the row)

  for (int ks = 0; ks < 16; ++ks) {
    // ---- B fragment: 16 contiguous fp32 -> f16 (K = ks*32 + hi*16 + 0..15)
    const int kb = ks * 32 + hi * 16;
    const float4 f0 = *(const float4*)(wrow + kb);
    const float4 f1 = *(const float4*)(wrow + kb + 4);
    const float4 f2 = *(const float4*)(wrow + kb + 8);
    const float4 f3 = *(const float4*)(wrow + kb + 12);
    float fv[16] = {f0.x, f0.y, f0.z, f0.w, f1.x, f1.y, f1.z, f1.w,
                    f2.x, f2.y, f2.z, f2.w, f3.x, f3.y, f3.z, f3.w};
    v16h b;
#pragma unroll
    for (int i = 0; i < 16; ++i) {
      const _Float16 h = (_Float16)fv[i];
      b[i] = h;
      const float hf = (float)h;
      ssq = fmaf(hf, hf, ssq);
    }

    // ---- 8 M-tiles: A fragment + WMMA
    const int ka = ks * 32 + hi * 16;  // lanes 0-15: K 0..15; 16-31: K 16..31
#pragma unroll
    for (int mt = 0; mt < 8; ++mt) {
      const int arow = rowbase + mt * 16 + l15;
      const _Float16* ap = e16 + (size_t)arow * EMB_DIM + ka;
      const v8h a0 = *(const v8h*)ap;          // K = ka .. ka+7
      const v8h a1 = *(const v8h*)(ap + 8);    // K = ka+8 .. ka+15
      v16h a;
#pragma unroll
      for (int i = 0; i < 8; ++i) { a[i] = a0[i]; a[8 + i] = a1[i]; }
      c[mt] = __builtin_amdgcn_wmma_f32_16x16x32_f16(
          false, a, false, b, (short)0, c[mt], false, false);
    }
  }

  // ---- epilogue: complete per-class sumsq, scale, clip, exp, row-reduce
  ssq += __shfl_xor(ssq, 16, 32);                       // pair lanes l, l+16
  const float rw = 1.0f / fmaxf(sqrtf(ssq), 1e-12f);    // 1/||f16(w_cls)||

  const int chunk = blockIdx.x * 2 + nw;                // 16-class chunk id
  float* pc = part + (size_t)chunk * BATCH;

#pragma unroll
  for (int mt = 0; mt < 8; ++mt) {
#pragma unroll
    for (int r = 0; r < 8; ++r) {
      float cs = c[mt][r] * rw;
      cs = fminf(fmaxf(cs, -CLIPV), CLIPV);
      float v = __expf(SSCALE * cs - M0_SHIFT);
      v += __shfl_xor(v, 1, 32);
      v += __shfl_xor(v, 2, 32);
      v += __shfl_xor(v, 4, 32);
      v += __shfl_xor(v, 8, 32);                        // sum over 16 classes
      if (l15 == 0) {
        const int row = rowbase + mt * 16 + r + hi * 8; // C layout: M = r + 8*hi
        pc[row] = v;
      }
    }
  }
}

// ---------------------------------------------------------------------- K2 --
__global__ __launch_bounds__(256)
void finalize_row_kernel(const float* __restrict__ part,
                         const _Float16* __restrict__ e16,
                         const float* __restrict__ w,
                         const int* __restrict__ labels,
                         float* __restrict__ nll) {
  __shared__ float sse[256], sd[256], sq[256];
  const int t   = threadIdx.x;
  const int row = blockIdx.x;

  float se = 0.0f;
  for (int ch = t; ch < NCHUNKS; ch += 256) se += part[(size_t)ch * BATCH + row];

  // label-column cosine, recomputed with the same f16 recipe as the GEMM
  const int lab = labels[row];
  const float*    wr = w  + (size_t)lab * EMB_DIM;
  const _Float16* er = e16 + (size_t)row * EMB_DIM;
  float d = 0.0f, q = 0.0f;
#pragma unroll
  for (int j = 0; j < 2; ++j) {
    const int k = t * 2 + j;
    const _Float16 h = (_Float16)wr[k];
    const float hf = (float)h;
    d = fmaf((float)er[k], hf, d);
    q = fmaf(hf, hf, q);
  }
  sse[t] = se; sd[t] = d; sq[t] = q;
  __syncthreads();
  for (int s = 128; s > 0; s >>= 1) {
    if (t < s) { sse[t] += sse[t + s]; sd[t] += sd[t + s]; sq[t] += sq[t + s]; }
    __syncthreads();
  }
  if (t == 0) {
    const float rw = 1.0f / fmaxf(sqrtf(sq[0]), 1e-12f);
    float cl = sd[0] * rw;
    cl = fminf(fmaxf(cl, -CLIPV), CLIPV);
    const float sine = sqrtf(fmaxf(1.0f - cl * cl, 0.0f));
    float phi = cl * ARC_COS_M - sine * ARC_SIN_M;
    if (!(cl > ARC_TH)) phi = cl - ARC_MM;
    const float lgl = SSCALE * cl;
    const float lgp = SSCALE * phi;
    // swap label logit: S*cos -> S*phi inside the (shifted) sum of exps
    const float seN = sse[0] - __expf(lgl - M0_SHIFT) + __expf(lgp - M0_SHIFT);
    nll[row] = logf(fmaxf(seN, 1e-30f)) + M0_SHIFT - lgp;
  }
}

// ---------------------------------------------------------------------- K3 --
__global__ __launch_bounds__(512)
void mean_kernel(const float* __restrict__ nll, float* __restrict__ out) {
  __shared__ float s[512];
  const int t = threadIdx.x;
  s[t] = nll[t];
  __syncthreads();
  for (int k = 256; k > 0; k >>= 1) {
    if (t < k) s[t] += s[t + k];
    __syncthreads();
  }
  if (t == 0) out[0] = s[0] * (1.0f / (float)BATCH);
}

// ---------------------------------------------------------------------------
extern "C" void kernel_launch(void* const* d_in, const int* in_sizes, int n_in,
                              void* d_out, int out_size, void* d_ws, size_t ws_size,
                              hipStream_t stream) {
  const float* emb    = (const float*)d_in[0];   // [512, 512]  f32
  const float* weight = (const float*)d_in[1];   // [100000, 512] f32
  const int*   labels = (const int*)d_in[2];     // [512] i32
  float* out = (float*)d_out;

  // workspace layout
  char* ws = (char*)d_ws;
  float*     part = (float*)ws;                                   // 6250*512 f32 = 12.8 MB
  const size_t part_bytes = (size_t)NCHUNKS * BATCH * sizeof(float);
  float*     nll  = (float*)(ws + part_bytes);                    // 512 f32
  _Float16*  e16  = (_Float16*)(ws + part_bytes + 2048);          // 512*512 f16 (2KB-aligned)

  norm_e_kernel     <<<BATCH / 8,   256, 0, stream>>>(emb, e16);
  arcface_gemm_kernel<<<GEMM_BLOCKS, 256, 0, stream>>>(weight, e16, part);
  finalize_row_kernel<<<BATCH,      256, 0, stream>>>(part, e16, weight, labels, nll);
  mean_kernel       <<<1,           512, 0, stream>>>(nll, out);
}
